// Attn_14345190768775
// MI455X (gfx1250) — matrix operand order repaired and checked by reading, not compiled, not run on hardware
//
#include <hip/hip_runtime.h>
#include <hip/hip_bf16.h>

#define S_   2048
#define H_   2048
#define NH_  16
#define NKV_ 8
#define HD_  128
#define TH_  1024

typedef __attribute__((ext_vector_type(16))) __bf16 v16bf;
typedef __attribute__((ext_vector_type(8)))  float  v8f;
typedef __attribute__((ext_vector_type(4)))  int    v4i;

#if defined(__has_builtin)
#  if __has_builtin(__builtin_amdgcn_global_load_async_to_lds_b128)
#    define USE_ASYNC_LDS 1
#  endif
#endif
#ifndef USE_ASYNC_LDS
#  define USE_ASYNC_LDS 0
#endif

#if USE_ASYNC_LDS
typedef __attribute__((address_space(1))) v4i g_v4i;
typedef __attribute__((address_space(3))) v4i l_v4i;
__device__ __forceinline__ void async_b128(const void* g, void* l) {
  __builtin_amdgcn_global_load_async_to_lds_b128((g_v4i*)g, (l_v4i*)l, 0, 0);
}
__device__ __forceinline__ void wait_async0() {
#if __has_builtin(__builtin_amdgcn_s_wait_asynccnt)
  __builtin_amdgcn_s_wait_asynccnt(0);
#else
  asm volatile("s_wait_asynccnt 0" ::: "memory");
#endif
}
#endif

union FragA {
  v16bf v;
  unsigned short h[16];
  unsigned int   u[8];
};

__device__ __forceinline__ unsigned short f2bf(float f) {
  union { float f; unsigned u; } c; c.f = f;
  unsigned r = c.u + 0x7FFFu + ((c.u >> 16) & 1u);   // round-to-nearest-even
  return (unsigned short)(r >> 16);
}

// A-matrix (16-bit, 16x32) dword(K-pair) index per ISA layout:
// lanes 0-15 hold K pairs {0..3, 8..11}; lanes 16-31 hold {4..7, 12..15}
__device__ __forceinline__ int a_kp(int j, int h) { return (j < 4) ? (j + 4*h) : (j + 4 + 4*h); }
// B-matrix (32x16) dword index: same K order as A (lanes 0-15 {0..3, 8..11}; lanes 16-31 {4..7, 12..15})
__device__ __forceinline__ int b_kp(int j, int h) { return (j < 4) ? (j + 4*h) : (j + 4 + 4*h); }

// ---------------------------------------------------------------------------
// Tiled GEMM: C(MxN, f32) = A(MxK, bf16) * B(NxK, bf16)^T
// block = 128 thr = 4 waves; block tile 128x64; wave tile 32x64; K step 32.
// ---------------------------------------------------------------------------
__global__ __launch_bounds__(128) void gemm_nt_bf16(
    const unsigned short* __restrict__ A, const unsigned short* __restrict__ B,
    float* __restrict__ C, int M, int N, int K)
{
  __shared__ __align__(16) unsigned int lA[128 * 16];  // 128 rows x 32 bf16
  __shared__ __align__(16) unsigned int lB[64 * 16];   //  64 rows x 32 bf16

  const int tid  = threadIdx.x;
  const int lane = tid & 31, wave = tid >> 5;
  const int h = lane >> 4, nl = lane & 15;
  const int m0 = blockIdx.y * 128, n0 = blockIdx.x * 64;

  const uint4* A4 = (const uint4*)A;
  const uint4* B4 = (const uint4*)B;
  const int Kw4 = K >> 3;   // uint4 per row

  v8f acc[2][4];
  for (int mi = 0; mi < 2; mi++)
    for (int t = 0; t < 4; t++)
      for (int r = 0; r < 8; r++) acc[mi][t][r] = 0.0f;

  for (int kk = 0; kk < K; kk += 32) {
    const int k4 = kk >> 3;
#if USE_ASYNC_LDS
    for (int i = 0; i < 4; i++) {          // A tile: 512 uint4
      int v = i * 128 + tid, row = v >> 2, c4 = v & 3;
      async_b128(&A4[(size_t)(m0 + row) * Kw4 + k4 + c4], &((uint4*)lA)[v]);
    }
    for (int i = 0; i < 2; i++) {          // B tile: 256 uint4
      int v = i * 128 + tid, row = v >> 2, c4 = v & 3;
      async_b128(&B4[(size_t)(n0 + row) * Kw4 + k4 + c4], &((uint4*)lB)[v]);
    }
    wait_async0();
#else
    for (int i = 0; i < 4; i++) {
      int v = i * 128 + tid, row = v >> 2, c4 = v & 3;
      ((uint4*)lA)[v] = A4[(size_t)(m0 + row) * Kw4 + k4 + c4];
    }
    for (int i = 0; i < 2; i++) {
      int v = i * 128 + tid, row = v >> 2, c4 = v & 3;
      ((uint4*)lB)[v] = B4[(size_t)(n0 + row) * Kw4 + k4 + c4];
    }
#endif
    if (kk + 32 < K) {   // global_prefetch_b8 hint for next K tile
      __builtin_prefetch(&A4[(size_t)(m0 + (tid >> 2)) * Kw4 + k4 + 4], 0, 1);
      __builtin_prefetch(&B4[(size_t)(n0 + (tid >> 2)) * Kw4 + k4 + 4], 0, 1);
    }
    __syncthreads();

    FragA a[2];
    for (int mi = 0; mi < 2; mi++) {
      int row = wave * 32 + mi * 16 + nl;
      for (int j = 0; j < 8; j++) a[mi].u[j] = lA[row * 16 + a_kp(j, h)];
    }
    for (int t = 0; t < 4; t++) {
      FragA b;
      int row = t * 16 + nl;
      for (int j = 0; j < 8; j++) b.u[j] = lB[row * 16 + b_kp(j, h)];
      acc[0][t] = __builtin_amdgcn_wmma_f32_16x16x32_bf16(
          false, a[0].v, false, b.v, (short)0, acc[0][t], false, false);
      acc[1][t] = __builtin_amdgcn_wmma_f32_16x16x32_bf16(
          false, a[1].v, false, b.v, (short)0, acc[1][t], false, false);
    }
    __syncthreads();
  }

  for (int mi = 0; mi < 2; mi++)
    for (int t = 0; t < 4; t++)
      for (int r = 0; r < 8; r++)
        C[(size_t)(m0 + wave * 32 + mi * 16 + r + 8 * h) * N + n0 + t * 16 + nl]
            = acc[mi][t][r];
}

// ---------------------------------------------------------------------------
// Flash attention (causal, GQA 16q/8kv, HD=128), bf16 WMMA, f32 accumulate.
// grid = (S/64, NH); block = 128 thr = 4 waves; wave owns 16 query rows.
// V is pre-transposed to (NKV*HD) x S so both K and V tiles stage coalesced.
// ---------------------------------------------------------------------------
__global__ __launch_bounds__(128) void flash_attn_k(
    const unsigned short* __restrict__ Q,   // S x NH  x HD  (post norm+rope)
    const unsigned short* __restrict__ Kk,  // S x NKV x HD
    const unsigned short* __restrict__ Vt,  // (NKV*HD) x S   transposed
    float* __restrict__ O)                  // S x (NH*HD) f32
{
  __shared__ __align__(16) unsigned int lK[32 * 64];    // 32 keys x 128 bf16
  __shared__ __align__(16) unsigned int lV[128 * 16];   // 128 feats x 32 keys
  __shared__ __align__(16) unsigned int lP[4 * 16 * 16];// per-wave 16x32 P

  const int tid = threadIdx.x, lane = tid & 31, wave = tid >> 5;
  const int h = lane >> 4, nl = lane & 15;
  const int head = blockIdx.y, kvh = head >> 1;
  const int q0 = blockIdx.x * 64 + wave * 16;

  const unsigned int* Qu = (const unsigned int*)Q;
  const uint4* K4 = (const uint4*)Kk;
  const uint4* V4 = (const uint4*)Vt;

  FragA qf[4];
  {
    size_t base = (size_t)(q0 + nl) * (NH_ * HD_ / 2) + head * (HD_ / 2);
    for (int f = 0; f < 4; f++)
      for (int j = 0; j < 8; j++)
        qf[f].u[j] = Qu[base + f * 16 + a_kp(j, h)];
  }

  v8f o[8];
  float m[8], l[8], al[8];
  for (int t = 0; t < 8; t++)
    for (int r = 0; r < 8; r++) o[t][r] = 0.0f;
  for (int r = 0; r < 8; r++) { m[r] = -3.0e38f; l[r] = 0.0f; }

  const float scale = 0.08838834764831845f;   // HD^-0.5
  const int nkb = (blockIdx.x * 64 + 64) / 32;

  for (int kb = 0; kb < nkb; kb++) {
#if USE_ASYNC_LDS
    for (int i = 0; i < 4; i++) {        // K tile: 512 uint4, key-major
      int v = i * 128 + tid, row = v >> 4, c4 = v & 15;
      async_b128(&K4[(size_t)(kb * 32 + row) * (NKV_ * HD_ / 8) + kvh * 16 + c4],
                 &((uint4*)lK)[v]);
    }
    for (int i = 0; i < 4; i++) {        // V tile: 512 uint4, feature-major
      int v = i * 128 + tid, row = v >> 2, c4 = v & 3;
      async_b128(&V4[(size_t)(kvh * HD_ + row) * (S_ / 8) + kb * 4 + c4],
                 &((uint4*)lV)[v]);
    }
    wait_async0();
#else
    for (int i = 0; i < 4; i++) {
      int v = i * 128 + tid, row = v >> 4, c4 = v & 15;
      ((uint4*)lK)[v] = K4[(size_t)(kb * 32 + row) * (NKV_ * HD_ / 8) + kvh * 16 + c4];
    }
    for (int i = 0; i < 4; i++) {
      int v = i * 128 + tid, row = v >> 2, c4 = v & 3;
      ((uint4*)lV)[v] = V4[(size_t)(kvh * HD_ + row) * (S_ / 8) + kb * 4 + c4];
    }
#endif
    __syncthreads();

    // S tiles: keys kb*32 + [0..15] and [16..31]
    v8f s0, s1;
    for (int r = 0; r < 8; r++) { s0[r] = 0.0f; s1[r] = 0.0f; }
    for (int f = 0; f < 4; f++) {
      FragA b0, b1;
      for (int j = 0; j < 8; j++) {
        b0.u[j] = lK[nl * 64 + f * 16 + b_kp(j, h)];
        b1.u[j] = lK[(16 + nl) * 64 + f * 16 + b_kp(j, h)];
      }
      s0 = __builtin_amdgcn_wmma_f32_16x16x32_bf16(false, qf[f].v, false, b0.v, (short)0, s0, false, false);
      s1 = __builtin_amdgcn_wmma_f32_16x16x32_bf16(false, qf[f].v, false, b1.v, (short)0, s1, false, false);
    }

    // online softmax (rows confined to 16-lane halves in C layout)
    unsigned short* Pw = (unsigned short*)&lP[wave * 16 * 16];
    const int k0g = kb * 32 + nl, k1g = k0g + 16;
    for (int r = 0; r < 8; r++) {
      int qg = q0 + r + 8 * h;
      float v0 = (k0g <= qg) ? s0[r] * scale : -3.0e38f;
      float v1 = (k1g <= qg) ? s1[r] * scale : -3.0e38f;
      float rm = fmaxf(v0, v1);
      for (int mm = 8; mm; mm >>= 1) rm = fmaxf(rm, __shfl_xor(rm, mm, 16));
      float mn = fmaxf(m[r], rm);
      float a0 = __expf(m[r] - mn);
      float p0 = __expf(v0 - mn), p1 = __expf(v1 - mn);
      float rs = p0 + p1;
      for (int mm = 8; mm; mm >>= 1) rs += __shfl_xor(rs, mm, 16);
      m[r] = mn; l[r] = l[r] * a0 + rs; al[r] = a0;
      Pw[(r + 8 * h) * 32 + nl]      = f2bf(p0);
      Pw[(r + 8 * h) * 32 + 16 + nl] = f2bf(p1);
    }
    for (int t = 0; t < 8; t++)
      for (int r = 0; r < 8; r++) o[t][r] *= al[r];

    // O += P(16x32) * V(32x128); P re-read in A layout via per-wave LDS tile
    FragA pf;
    {
      const unsigned int* Pu = &lP[wave * 16 * 16];
      for (int j = 0; j < 8; j++) pf.u[j] = Pu[nl * 16 + a_kp(j, h)];
    }
    for (int t = 0; t < 8; t++) {
      FragA vf;
      int d = t * 16 + nl;                  // output feature (B-matrix column)
      for (int j = 0; j < 8; j++) vf.u[j] = lV[d * 16 + b_kp(j, h)];
      o[t] = __builtin_amdgcn_wmma_f32_16x16x32_bf16(false, pf.v, false, vf.v, (short)0, o[t], false, false);
    }
    __syncthreads();
  }

  for (int r = 0; r < 8; r++) {
    float inv = 1.0f / l[r];
    int qg = q0 + r + 8 * h;
    for (int t = 0; t < 8; t++)
      O[(size_t)qg * (NH_ * HD_) + head * HD_ + t * 16 + nl] = o[t][r] * inv;
  }
}

// ---------------------------------------------------------------------------
// Fused RMSNorm (over HD=128) + RoPE, f32 in -> bf16 out. One block per row.
// ---------------------------------------------------------------------------
__global__ __launch_bounds__(128) void rmsnorm_rope_k(
    const float* __restrict__ X, const float* __restrict__ cosT,
    const float* __restrict__ sinT, const float* __restrict__ w,
    unsigned short* __restrict__ Y, int nheads)
{
  __shared__ float xs[128];
  __shared__ float red[128];
  const int row = blockIdx.x;            // s * nheads + head
  const int s = row / nheads;
  const int d = threadIdx.x;

  float x = X[(size_t)row * 128 + d];
  red[d] = x * x;
  __syncthreads();
  for (int st = 64; st > 0; st >>= 1) {
    if (d < st) red[d] += red[d + st];
    __syncthreads();
  }
  float r = rsqrtf(red[0] * (1.0f / 128.0f) + 1e-6f);
  float xn = x * r * w[d];
  xs[d] = xn;
  __syncthreads();
  float c  = cosT[(size_t)s * 128 + d];
  float sn = sinT[(size_t)s * 128 + d];
  float rot = (d < 64) ? -xs[d + 64] : xs[d - 64];
  Y[(size_t)row * 128 + d] = f2bf(xn * c + rot * sn);
}

__global__ void f32_to_bf16_k(const float* __restrict__ X,
                              unsigned short* __restrict__ Y, size_t n)
{
  size_t i = (size_t)blockIdx.x * blockDim.x + threadIdx.x;
  if (i < n) Y[i] = f2bf(X[i]);
}

// Vf: S x (NKV*HD) f32  ->  Vt: (NKV*HD) x S bf16 (coalesced writes)
__global__ void v_transpose_k(const float* __restrict__ Vf,
                              unsigned short* __restrict__ Vt)
{
  size_t i = (size_t)blockIdx.x * blockDim.x + threadIdx.x;
  if (i >= (size_t)NKV_ * HD_ * S_) return;
  int s = (int)(i % S_);
  int d = (int)(i / S_);
  Vt[i] = f2bf(Vf[(size_t)s * (NKV_ * HD_) + d]);
}

__global__ void mse_partial_k(const float* __restrict__ a, const float* __restrict__ b,
                              size_t n, float* __restrict__ acc)
{
  __shared__ float red[256];
  size_t i0 = (size_t)blockIdx.x * blockDim.x + threadIdx.x;
  float s = 0.0f;
  for (size_t i = i0; i < n; i += (size_t)gridDim.x * blockDim.x) {
    float d = a[i] - b[i];
    s += d * d;
  }
  red[threadIdx.x] = s;
  __syncthreads();
  for (int st = 128; st > 0; st >>= 1) {
    if ((int)threadIdx.x < st) red[threadIdx.x] += red[threadIdx.x + st];
    __syncthreads();
  }
  if (threadIdx.x == 0) atomicAdd(acc, red[0]);
}

__global__ void zero2_k(float* p) { if (threadIdx.x < 2) p[threadIdx.x] = 0.0f; }

__global__ void finalize_loss_k(const float* acc, float* out, float invn) {
  if (threadIdx.x == 0) *out = *acc * invn;
}

// ---------------------------------------------------------------------------
extern "C" void kernel_launch(void* const* d_in, const int* in_sizes, int n_in,
                              void* d_out, int out_size, void* d_ws, size_t ws_size,
                              hipStream_t stream)
{
  (void)in_sizes; (void)n_in; (void)out_size; (void)ws_size;

  const float* hs   = (const float*)d_in[0];
  const float* chs  = (const float*)d_in[1];
  const float* cosT = (const float*)d_in[3];
  const float* sinT = (const float*)d_in[4];
  const float* Wq   = (const float*)d_in[5];
  const float* Wk   = (const float*)d_in[6];
  const float* Wv   = (const float*)d_in[7];
  const float* Wo   = (const float*)d_in[8];
  const float* qn   = (const float*)d_in[9];
  const float* kn   = (const float*)d_in[10];
  const float* Wup  = (const float*)d_in[11];
  const float* Wdn  = (const float*)d_in[12];

  char* wsp = (char*)d_ws;
  size_t off = 0;
  auto carve = [&](size_t bytes) -> void* {
    void* p = wsp + off;
    off = (off + bytes + 255) & ~(size_t)255;
    return p;
  };

  unsigned short* Wq_b  = (unsigned short*)carve((size_t)H_ * H_ * 2);
  unsigned short* Wk_b  = (unsigned short*)carve((size_t)NKV_ * HD_ * H_ * 2);
  unsigned short* Wv_b  = (unsigned short*)carve((size_t)NKV_ * HD_ * H_ * 2);
  unsigned short* Wo_b  = (unsigned short*)carve((size_t)H_ * H_ * 2);
  unsigned short* Wup_b = (unsigned short*)carve((size_t)H_ * TH_ * 2);
  unsigned short* Wdn_b = (unsigned short*)carve((size_t)TH_ * H_ * 2);
  unsigned short* Xb    = (unsigned short*)carve((size_t)S_ * H_ * 2);
  float*          Qf    = (float*)carve((size_t)S_ * H_ * 4);
  float*          Kf    = (float*)carve((size_t)S_ * NKV_ * HD_ * 4);
  float*          Vf    = (float*)carve((size_t)S_ * NKV_ * HD_ * 4);
  unsigned short* Qb    = (unsigned short*)carve((size_t)S_ * H_ * 2);
  unsigned short* Kb    = (unsigned short*)carve((size_t)S_ * NKV_ * HD_ * 2);
  unsigned short* Vtb   = (unsigned short*)carve((size_t)NKV_ * HD_ * S_ * 2);
  float*          At    = (float*)carve((size_t)S_ * H_ * 4);
  unsigned short* Ab    = (unsigned short*)carve((size_t)S_ * H_ * 2);
  float*          zoom  = (float*)carve((size_t)S_ * H_ * 4);
  float*          out2  = (float*)carve((size_t)S_ * H_ * 4);
  unsigned short* out2b = (unsigned short*)carve((size_t)S_ * H_ * 2);
  unsigned short* chsb  = (unsigned short*)carve((size_t)S_ * TH_ * 2);
  float*          acc   = (float*)carve(256);

  auto cvt = [&](const float* x, unsigned short* y, size_t n) {
    f32_to_bf16_k<<<(unsigned)((n + 255) / 256), 256, 0, stream>>>(x, y, n);
  };
  auto gemm = [&](const unsigned short* A, const unsigned short* Bm, float* C,
                  int M, int N, int K) {
    gemm_nt_bf16<<<dim3(N / 64, M / 128), 128, 0, stream>>>(A, Bm, C, M, N, K);
  };

  // weight conversions (deterministic, redone per call)
  cvt(Wq,  Wq_b,  (size_t)H_ * H_);
  cvt(Wk,  Wk_b,  (size_t)NKV_ * HD_ * H_);
  cvt(Wv,  Wv_b,  (size_t)NKV_ * HD_ * H_);
  cvt(Wo,  Wo_b,  (size_t)H_ * H_);
  cvt(Wup, Wup_b, (size_t)H_ * TH_);
  cvt(Wdn, Wdn_b, (size_t)TH_ * H_);

  auto branch = [&](const float* X, float* Oproj) {
    cvt(X, Xb, (size_t)S_ * H_);
    gemm(Xb, Wq_b, Qf, S_, H_, H_);
    gemm(Xb, Wk_b, Kf, S_, NKV_ * HD_, H_);
    gemm(Xb, Wv_b, Vf, S_, NKV_ * HD_, H_);
    rmsnorm_rope_k<<<S_ * NH_, 128, 0, stream>>>(Qf, cosT, sinT, qn, Qb, NH_);
    rmsnorm_rope_k<<<S_ * NKV_, 128, 0, stream>>>(Kf, cosT, sinT, kn, Kb, NKV_);
    v_transpose_k<<<(unsigned)(((size_t)NKV_ * HD_ * S_ + 255) / 256), 256, 0, stream>>>(Vf, Vtb);
    flash_attn_k<<<dim3(S_ / 64, NH_), 128, 0, stream>>>(Qb, Kb, Vtb, At);
    cvt(At, Ab, (size_t)S_ * H_);
    gemm(Ab, Wo_b, Oproj, S_, H_, H_);
  };

  float* raw_out  = (float*)d_out;
  float* comp_out = raw_out + (size_t)S_ * H_;
  float* loss_in  = comp_out + (size_t)S_ * TH_;
  float* loss_out = loss_in + 1;

  // branch 1: raw
  branch(hs, raw_out);

  // up-projection + branch 2 + down-projection
  cvt(chs, chsb, (size_t)S_ * TH_);
  gemm(chsb, Wup_b, zoom, S_, H_, TH_);
  branch(zoom, out2);
  cvt(out2, out2b, (size_t)S_ * H_);
  gemm(out2b, Wdn_b, comp_out, S_, TH_, H_);

  // losses
  zero2_k<<<1, 32, 0, stream>>>(acc);
  mse_partial_k<<<1024, 256, 0, stream>>>(zoom, hs, (size_t)S_ * H_, acc);
  mse_partial_k<<<1024, 256, 0, stream>>>(out2, raw_out, (size_t)S_ * H_, acc + 1);
  const float invn = 1.0f / ((float)S_ * (float)H_);
  finalize_loss_k<<<1, 32, 0, stream>>>(acc,     loss_in,  invn);
  finalize_loss_k<<<1, 32, 0, stream>>>(acc + 1, loss_out, invn);
}
